// MultiHeadAttention_33071248179700
// MI455X (gfx1250) — matrix operand order repaired and checked by reading, not compiled, not run on hardware
//
#include <hip/hip_runtime.h>

// MI455X / gfx1250 multi-head attention, bf16 WMMA pipeline + double-buffered
// async-to-LDS. B=4, L=2048, D=1024, H=16, DK=64.

typedef unsigned short u16;
typedef unsigned int   u32;
typedef unsigned long long u64;

typedef __bf16 bf16;
typedef bf16  v16bf __attribute__((ext_vector_type(16)));
typedef float v8f   __attribute__((ext_vector_type(8)));

union Frag16 {
    v16bf v;
    uint4 q[2];
};

__device__ inline u16 f2bf(float x) {
    union { bf16 h; u16 u; } c;
    c.h = (bf16)x;
    return c.u;
}
__device__ inline u32 pk2bf(float a, float b) {
    union { bf16 h[2]; u32 u; } c;
    c.h[0] = (bf16)a; c.h[1] = (bf16)b;
    return c.u;
}
__device__ inline u32 lds_off(const void* p) {   // generic LDS ptr -> 32-bit LDS offset
    return (u32)(size_t)p;
}
__device__ inline float exp2f_hw(float x) {      // native v_exp_f32 (exp2)
    return __builtin_amdgcn_exp2f(x);
}

// ---------------------------------------------------------------------------
// Projection GEMM: Y[M,N] = X[M,K] @ W[N,K]^T + bias,  M=8192, N=K=1024.
// Block tile 128(M) x 64(N), 8 waves (4x2), wave tile 32x32, K-step 64.
// OUT_MODE: 0 = f32 row-major [M,N] (final output)
//           1 = bf16 head-major [B,H,L,DK]        (Q, K)
//           2 = bf16 transposed head-major [B,H,DK,L], packed b128 (V)
// ---------------------------------------------------------------------------
template <bool X_IS_BF16, int OUT_MODE>
__global__ __launch_bounds__(256) void proj_gemm(const void* __restrict__ Xv,
                                                 const float* __restrict__ W,
                                                 const float* __restrict__ bias,
                                                 void* __restrict__ Yv) {
    constexpr int K  = 1024;
    constexpr int TS = 72;   // LDS row stride (ushorts) = 36 dwords -> conflict-free
    __shared__ __align__(16) u16 At[128 * TS];
    __shared__ __align__(16) u16 Bt[64 * TS];

    const int tid  = threadIdx.x;
    const int w    = tid >> 5;
    const int lane = tid & 31;
    const int n    = lane & 15;
    const int hi   = lane >> 4;
    const int m0   = blockIdx.y * 128;
    const int n0   = blockIdx.x * 64;
    const int wm   = (w >> 1) * 32;
    const int wn   = (w & 1) * 32;

    v8f acc[2][2] = {};

    for (int k0 = 0; k0 < K; k0 += 64) {
        __syncthreads();
        // Stage A tile: 128 rows x 64 k as bf16 (2048 4-elem chunks).
        #pragma unroll
        for (int c = tid; c < 2048; c += 256) {
            const int row = c >> 4, k4 = (c & 15) * 4;
            u32 lo, hic;
            if (X_IS_BF16) {
                const u16* X = (const u16*)Xv;
                const u16* p = X + (size_t)(m0 + row) * K + k0 + k4;
                uint2 u = *(const uint2*)p;
                lo = u.x; hic = u.y;
                if (k0 + 64 < K) __builtin_prefetch(p + 64, 0, 3);
            } else {
                const float* X = (const float*)Xv;
                const float* p = X + (size_t)(m0 + row) * K + k0 + k4;
                float4 f = *(const float4*)p;
                lo  = pk2bf(f.x, f.y);
                hic = pk2bf(f.z, f.w);
                if (k0 + 64 < K) __builtin_prefetch(p + 64, 0, 3);
            }
            uint2 st; st.x = lo; st.y = hic;
            *(uint2*)&At[row * TS + k4] = st;
        }
        // Stage B tile: 64 rows (N) x 64 k from W[N,K] (1024 chunks).
        #pragma unroll
        for (int c = tid; c < 1024; c += 256) {
            const int row = c >> 4, k4 = (c & 15) * 4;
            const float* p = W + (size_t)(n0 + row) * K + k0 + k4;
            float4 f = *(const float4*)p;
            uint2 st; st.x = pk2bf(f.x, f.y); st.y = pk2bf(f.z, f.w);
            *(uint2*)&Bt[row * TS + k4] = st;
            if (k0 + 64 < K) __builtin_prefetch(p + 64, 0, 3);
        }
        __syncthreads();

        #pragma unroll
        for (int hf = 0; hf < 2; ++hf) {
            Frag16 a[2], b[2];
            #pragma unroll
            for (int tm = 0; tm < 2; ++tm) {
                // A layout: lane<16 -> K {0..7,16..23}; lane>=16 -> K {8..15,24..31}
                const u16* p = &At[(wm + tm * 16 + n) * TS + hf * 32 + hi * 8];
                a[tm].q[0] = *(const uint4*)p;
                a[tm].q[1] = *(const uint4*)(p + 16);
            }
            #pragma unroll
            for (int tn = 0; tn < 2; ++tn) {
                // B layout: lane<16 -> K {0..7,16..23}; lane>=16 -> K {8..15,24..31}
                const u16* p = &Bt[(wn + tn * 16 + n) * TS + hf * 32 + hi * 8];
                b[tn].q[0] = *(const uint4*)p;
                b[tn].q[1] = *(const uint4*)(p + 16);
            }
            #pragma unroll
            for (int tm = 0; tm < 2; ++tm)
                #pragma unroll
                for (int tn = 0; tn < 2; ++tn)
                    acc[tm][tn] = __builtin_amdgcn_wmma_f32_16x16x32_bf16(
                        false, a[tm].v, false, b[tn].v, (short)0, acc[tm][tn],
                        false, false);
        }
    }

    // Epilogue. C/D layout: VGPR r -> row r + 8*hi, col = lane&15.
    #pragma unroll
    for (int tm = 0; tm < 2; ++tm) {
        #pragma unroll
        for (int tn = 0; tn < 2; ++tn) {
            const int colg = n0 + wn + tn * 16 + n;
            const float bv = bias[colg];
            if (OUT_MODE == 2) {
                // 8 consecutive rows per lane -> one packed b128 store.
                uint4 pk;
                pk.x = pk2bf(acc[tm][tn][0] + bv, acc[tm][tn][1] + bv);
                pk.y = pk2bf(acc[tm][tn][2] + bv, acc[tm][tn][3] + bv);
                pk.z = pk2bf(acc[tm][tn][4] + bv, acc[tm][tn][5] + bv);
                pk.w = pk2bf(acc[tm][tn][6] + bv, acc[tm][tn][7] + bv);
                const int rowb = m0 + wm + tm * 16 + hi * 8;
                const int bb = rowb >> 11, l = rowb & 2047;
                const int h = colg >> 6, dk = colg & 63;
                u16* Y = (u16*)Yv;
                *(uint4*)(Y + ((size_t)(bb * 16 + h) * 64 + dk) * 2048 + l) = pk;
            } else {
                #pragma unroll
                for (int r = 0; r < 8; ++r) {
                    const int rowg = m0 + wm + tm * 16 + r + hi * 8;
                    const float val = acc[tm][tn][r] + bv;
                    if (OUT_MODE == 1) {
                        u16* Y = (u16*)Yv;
                        const int bb = rowg >> 11, l = rowg & 2047;
                        const int h = colg >> 6, dk = colg & 63;
                        Y[(((size_t)(bb * 16 + h) * 2048 + l) << 6) + dk] = f2bf(val);
                    } else {
                        float* Y = (float*)Yv;
                        Y[(size_t)rowg * 1024 + colg] = val;
                    }
                }
            }
        }
    }
}

// ---------------------------------------------------------------------------
// Flash attention: block = (b,h) x 128 query rows; 8 waves x 16 rows.
// KV tiles of 32 staged via DOUBLE-BUFFERED global_load_async_to_lds_b128:
// tile j+1 copies (ASYNCcnt) overlap the 9 WMMAs + softmax of tile j;
// s_wait_asynccnt 2 retires exactly the previous tile's copies.
// Online softmax in exp2 domain; row-sum "l" via WMMA against all-ones B.
// Row-max reduction is level-major: 8 independent bpermutes per XOR level.
// q/k bf16 [B,H,L,DK]; v bf16 TRANSPOSED [B,H,DK,L]; ctx bf16 [B,L,H*DK].
// ---------------------------------------------------------------------------
__global__ __launch_bounds__(256) void flash_attn(const u16* __restrict__ q,
                                                  const u16* __restrict__ k,
                                                  const u16* __restrict__ vT,
                                                  const int* __restrict__ mask,
                                                  u16* __restrict__ ctx) {
    constexpr int KS  = 72;        // K tile stride (36 dwords, conflict-free)
    constexpr int VS  = 40;        // Vt / P tile stride (20 dwords, conflict-free)
    constexpr int KSZ = 32 * KS;   // one K buffer (ushorts)
    constexpr int VSZ = 64 * VS;   // one Vt buffer (ushorts)
    __shared__ __align__(16) u16 Kt[2 * KSZ];
    __shared__ __align__(16) u16 Vt[2 * VSZ];
    __shared__ __align__(16) u16 Pt[8 * 16 * VS];
    __shared__ int MaskL[2048];

    const int tid  = threadIdx.x;
    const int w    = tid >> 5;
    const int lane = tid & 31;
    const int n    = lane & 15;
    const int hi   = lane >> 4;
    const int bh   = blockIdx.y;       // b*H + h
    const int b    = bh >> 4;
    const int h    = bh & 15;
    const int qr   = blockIdx.x * 128 + w * 16;
    const size_t base  = (size_t)bh * 2048 * 64;   // [bh][l][dk]
    const size_t baseT = (size_t)bh * 64 * 2048;   // [bh][dk][l]

    // Stage this batch's mask row once (visible after first barrier).
    {
        const int4* src = (const int4*)(mask + b * 2048);
        int4* dst = (int4*)MaskL;
        for (int c = tid; c < 512; c += 256) dst[c] = src[c];
    }

    // Resident Q fragments (A layout, 16 rows x 64 dk).
    Frag16 qa[2];
    {
        const u16* qp = q + base + (size_t)(qr + n) * 64;
        qa[0].q[0] = *(const uint4*)(qp + hi * 8);
        qa[0].q[1] = *(const uint4*)(qp + 16 + hi * 8);
        qa[1].q[0] = *(const uint4*)(qp + 32 + hi * 8);
        qa[1].q[1] = *(const uint4*)(qp + 48 + hi * 8);
    }

    // Constant all-ones B fragment (bf16 1.0 = 0x3F80) for the row-sum WMMA.
    Frag16 ones;
    ones.q[0].x = ones.q[0].y = ones.q[0].z = ones.q[0].w = 0x3F803F80u;
    ones.q[1] = ones.q[0];

    v8f oacc[4] = {};
    v8f lacc    = {};
    float mrow[8];
    #pragma unroll
    for (int r = 0; r < 8; ++r) mrow[r] = -INFINITY;
    const float c2 = 0.125f * 1.44269504088896340736f;  // scale * log2(e)

    // Per-thread async-copy coordinates.
    const int krow = tid >> 3, kc8 = (tid & 7) * 8;   // K tile: 32 x 64
    const int vrow = tid >> 2, vc8 = (tid & 3) * 8;   // Vt tile: 64 x 32
    const u32 ldsK[2] = { lds_off(&Kt[krow * KS + kc8]),
                          lds_off(&Kt[KSZ + krow * KS + kc8]) };
    const u32 ldsV[2] = { lds_off(&Vt[vrow * VS + vc8]),
                          lds_off(&Vt[VSZ + vrow * VS + vc8]) };
    const u16* gKb = k  + base  + (size_t)krow * 64 + kc8;
    const u16* gVb = vT + baseT + (size_t)vrow * 2048 + vc8;

    // Prologue: start copying tile 0 into buffer 0.
    {
        const u64 gK = (u64)gKb;
        const u64 gV = (u64)gVb;
        asm volatile("global_load_async_to_lds_b128 %0, %1, off"
                     :: "v"(ldsK[0]), "v"(gK) : "memory");
        asm volatile("global_load_async_to_lds_b128 %0, %1, off"
                     :: "v"(ldsV[0]), "v"(gV) : "memory");
    }

    for (int jt = 0; jt < 64; ++jt) {
        const int cur = jt & 1, nxt = cur ^ 1;
        const int j0 = jt * 32;
        // Kick off tile jt+1 into the other buffer (clamped redundant copy on
        // the last iteration; retired by s_endpgm's implicit wait-idle).
        {
            const int jn = (jt + 1 < 64) ? (jt + 1) * 32 : j0;
            const u64 gK = (u64)(gKb + (size_t)jn * 64);
            const u64 gV = (u64)(gVb + jn);
            asm volatile("global_load_async_to_lds_b128 %0, %1, off"
                         :: "v"(ldsK[nxt]), "v"(gK) : "memory");
            asm volatile("global_load_async_to_lds_b128 %0, %1, off"
                         :: "v"(ldsV[nxt]), "v"(gV) : "memory");
        }
        // ASYNCcnt <= 2  ==> only the two just-issued remain; tile jt is in LDS.
        asm volatile("s_wait_asynccnt 0x2" ::: "memory");
        __syncthreads();

        const u16* Ktc = Kt + cur * KSZ;
        const u16* Vtc = Vt + cur * VSZ;

        // S = Q @ K^T for two 16-column tiles (contraction over dk = 2 x 32).
        v8f s0 = {}, s1 = {};
        #pragma unroll
        for (int hf = 0; hf < 2; ++hf) {
            Frag16 kb;
            const u16* p0 = &Ktc[n * KS + hf * 32 + hi * 8];
            kb.q[0] = *(const uint4*)p0;
            kb.q[1] = *(const uint4*)(p0 + 16);
            s0 = __builtin_amdgcn_wmma_f32_16x16x32_bf16(
                false, qa[hf].v, false, kb.v, (short)0, s0, false, false);
            const u16* p1 = &Ktc[(16 + n) * KS + hf * 32 + hi * 8];
            kb.q[0] = *(const uint4*)p1;
            kb.q[1] = *(const uint4*)(p1 + 16);
            s1 = __builtin_amdgcn_wmma_f32_16x16x32_bf16(
                false, qa[hf].v, false, kb.v, (short)0, s1, false, false);
        }

        // Masked, scaled scores in the exp2 domain: a = s*c2 + bias.
        const float bias0 = (MaskL[j0 + n]      != 0) ? 0.f : -1.0e9f;
        const float bias1 = (MaskL[j0 + 16 + n] != 0) ? 0.f : -1.0e9f;

        float a0[8], a1[8], t[8];
        #pragma unroll
        for (int r = 0; r < 8; ++r) {
            a0[r] = fmaf(s0[r], c2, bias0);
            a1[r] = fmaf(s1[r], c2, bias1);
            t[r]  = fmaxf(a0[r], a1[r]);
        }
        // Level-major row-max reduction across the 16 lanes of each half-wave:
        // 8 independent bpermutes in flight per XOR level.
        #pragma unroll
        for (int lvl = 1; lvl <= 8; lvl <<= 1) {
            float u[8];
            #pragma unroll
            for (int r = 0; r < 8; ++r) u[r] = __shfl_xor(t[r], lvl, 32);
            #pragma unroll
            for (int r = 0; r < 8; ++r) t[r] = fmaxf(t[r], u[r]);
        }

        u16* Pw = &Pt[w * 16 * VS];
        #pragma unroll
        for (int r = 0; r < 8; ++r) {
            const float mnew = fmaxf(mrow[r], t[r]);
            const float f = exp2f_hw(mrow[r] - mnew);
            lacc[r] *= f;
            #pragma unroll
            for (int g = 0; g < 4; ++g) oacc[g][r] *= f;
            const float e0 = exp2f_hw(a0[r] - mnew);
            const float e1 = exp2f_hw(a1[r] - mnew);
            mrow[r] = mnew;
            const int row = r + hi * 8;
            Pw[row * VS + n]      = f2bf(e0);
            Pw[row * VS + 16 + n] = f2bf(e1);
        }

        // Re-layout P (C/D -> A) through per-wave LDS (same-wave, in-order).
        Frag16 pa;
        pa.q[0] = *(const uint4*)&Pw[n * VS + hi * 8];
        pa.q[1] = *(const uint4*)&Pw[n * VS + 16 + hi * 8];

        // l += P @ ones  (row sums via the matrix pipe, no lane reduction).
        lacc = __builtin_amdgcn_wmma_f32_16x16x32_bf16(
            false, pa.v, false, ones.v, (short)0, lacc, false, false);

        // O += P @ V ; four 16-wide dk tiles from transposed Vt.
        #pragma unroll
        for (int g = 0; g < 4; ++g) {
            Frag16 vb;
            const u16* vp = &Vtc[(g * 16 + n) * VS + hi * 8];
            vb.q[0] = *(const uint4*)vp;
            vb.q[1] = *(const uint4*)(vp + 16);
            oacc[g] = __builtin_amdgcn_wmma_f32_16x16x32_bf16(
                false, pa.v, false, vb.v, (short)0, oacc[g], false, false);
        }
        __syncthreads();   // all reads of `cur` done; next iter may overwrite it
    }

    // Normalize and write context bf16, layout [B, L, H*DK].
    #pragma unroll
    for (int r = 0; r < 8; ++r) {
        const float inv = 1.0f / lacc[r];
        const int rowg = qr + r + hi * 8;
        const size_t obase = ((size_t)(b * 2048 + rowg)) * 1024 + h * 64;
        #pragma unroll
        for (int g = 0; g < 4; ++g)
            ctx[obase + g * 16 + n] = f2bf(oacc[g][r] * inv);
    }
}

// ---------------------------------------------------------------------------
extern "C" void kernel_launch(void* const* d_in, const int* in_sizes, int n_in,
                              void* d_out, int out_size, void* d_ws, size_t ws_size,
                              hipStream_t stream) {
    const float* query = (const float*)d_in[0];
    const float* key_  = (const float*)d_in[1];
    const float* value = (const float*)d_in[2];
    const int*   mask  = (const int*)d_in[3];
    const float* Wq = (const float*)d_in[4];
    const float* bq = (const float*)d_in[5];
    const float* Wk = (const float*)d_in[6];
    const float* bk = (const float*)d_in[7];
    const float* Wv = (const float*)d_in[8];
    const float* bv = (const float*)d_in[9];
    const float* Wo = (const float*)d_in[10];
    const float* bo = (const float*)d_in[11];

    const size_t NE = (size_t)4 * 16 * 2048 * 64;  // B*H*L*DK = 8,388,608
    u16* q_ws = (u16*)d_ws;
    u16* k_ws = q_ws + NE;
    u16* v_ws = k_ws + NE;   // V stored transposed: [B,H,DK,L]
    u16* c_ws = v_ws + NE;   // context bf16 [B,L,D]

    dim3 blk(256);
    dim3 gproj(1024 / 64, 8192 / 128);   // (16, 64)
    dim3 gattn(2048 / 128, 4 * 16);      // (16, 64)

    proj_gemm<false, 1><<<gproj, blk, 0, stream>>>(query, Wq, bq, q_ws);
    proj_gemm<false, 1><<<gproj, blk, 0, stream>>>(key_,  Wk, bk, k_ws);
    proj_gemm<false, 2><<<gproj, blk, 0, stream>>>(value, Wv, bv, v_ws);
    flash_attn<<<gattn, blk, 0, stream>>>(q_ws, k_ws, v_ws, mask, c_ws);
    proj_gemm<true, 0><<<gproj, blk, 0, stream>>>(c_ws, Wo, bo, d_out);
}
